// GCN_63831803953156
// MI455X (gfx1250) — matrix operand order repaired and checked by reading, not compiled, not run on hardware
//
#include <hip/hip_runtime.h>

typedef __bf16 bf16_t;
typedef __attribute__((ext_vector_type(16))) __bf16 v16bf;
typedef __attribute__((ext_vector_type(8)))  float  v8f;

#define GCN_N_NODES 50000
#define GCN_N_EDGES 600000
#define GCN_HID     128
#define GCN_OUT     64

// ---------------------------------------------------------------- utilities

__device__ __forceinline__ void atomic_add_f32(float* p, float v) {
  __hip_atomic_fetch_add(p, v, __ATOMIC_RELAXED, __HIP_MEMORY_SCOPE_AGENT);
}

__global__ void zero_f(float* __restrict__ p, int n) {
  int i = blockIdx.x * blockDim.x + threadIdx.x;
  if (i < n) p[i] = 0.0f;
}

__global__ void degree_kernel(const int* __restrict__ src, const int* __restrict__ dst,
                              float* __restrict__ deg_out, float* __restrict__ deg_in, int nE) {
  int e = blockIdx.x * blockDim.x + threadIdx.x;
  if (e < nE) {
    atomic_add_f32(&deg_out[src[e]], 1.0f);
    atomic_add_f32(&deg_in[dst[e]], 1.0f);
  }
}

__global__ void make_norms(float* __restrict__ p, int n) {
  int i = blockIdx.x * blockDim.x + threadIdx.x;
  if (i < n) p[i] = rsqrtf(fmaxf(p[i], 1.0f));
}

// A-fragment element index -> K offset within a 32-wide K tile (per ISA
// 16-bit A layout: VGPR j holds K pairs {0..7,16..23} (+8 for lanes 16-31)).
__device__ __forceinline__ int a_koff(int i) { return ((i & 8) ? 16 : 0) + (i & 7); }

// ------------------------------------------------- operand packing (bf16)

// apack[((mt*4 + kt)*32 + lane)*16 + i] = bf16( x[m][k] * out_norm[m] )
// with m = mt*16 + (lane&15), k = kt*32 + (lane>>4)*8 + a_koff(i).  Din==128.
__global__ void pack_x(const float* __restrict__ x, const float* __restrict__ out_norm,
                       bf16_t* __restrict__ apack, int total) {
  int idx = blockIdx.x * blockDim.x + threadIdx.x;
  if (idx >= total) return;
  int i    = idx & 15;
  int lane = (idx >> 4) & 31;
  int t    = idx >> 9;            // mt*4 + kt  (Ktiles = 128/32 = 4)
  int kt   = t & 3;
  int mt   = t >> 2;
  int m = mt * 16 + (lane & 15);
  int k = kt * 32 + (lane >> 4) * 8 + a_koff(i);
  apack[idx] = (bf16_t)(x[(size_t)m * 128 + k] * out_norm[m]);
}

// wpack[((kt*Ntiles + nt)*32 + lane)*16 + i] = bf16( W[k][n] )
// with n = nt*16 + (lane&15), k = kt*32 + (lane>>4)*8 + a_koff(i) (B-fragment layout).
__global__ void pack_w(const float* __restrict__ W, bf16_t* __restrict__ wpack,
                       int Dout, int total) {
  int idx = blockIdx.x * blockDim.x + threadIdx.x;
  if (idx >= total) return;
  int i    = idx & 15;
  int lane = (idx >> 4) & 31;
  int t    = idx >> 9;            // kt*Ntiles + nt
  int Ntiles = Dout >> 4;
  int nt = t % Ntiles;
  int kt = t / Ntiles;
  int n = nt * 16 + (lane & 15);
  int k = kt * 32 + (lane >> 4) * 8 + a_koff(i);
  wpack[idx] = (bf16_t)W[(size_t)k * Dout + n];
}

// ---------------------------------------------------------------- GEMM

// C[M x (NT*16)] = A[M x 128] * W[128 x (NT*16)], operands pre-packed in
// fragment order.  One wave per 16-row M-strip: the A fragment is loaded once
// per K-tile and reused across all NT column tiles held in NT accumulators
// (NT*8 VGPRs).  32 (NT=8) / 16 (NT=4) v_wmma_f32_16x16x32_bf16 per wave.
template <int NT>
__global__ void gemm_wmma_strip(const bf16_t* __restrict__ apack,
                                const bf16_t* __restrict__ wpack,
                                float* __restrict__ C, int Mtiles) {
  int mt = blockIdx.x * (blockDim.x >> 5) + (threadIdx.x >> 5);
  if (mt >= Mtiles) return;                 // wave-uniform: EXEC stays all-ones
  int lane = threadIdx.x & 31;

  const v16bf* aptr = (const v16bf*)apack + (size_t)mt * 4 * 32 + lane;
  const v16bf* bptr = (const v16bf*)wpack + lane;

  v8f acc[NT];
  v8f zero = {};
#pragma unroll
  for (int nt = 0; nt < NT; ++nt) acc[nt] = zero;

#pragma unroll
  for (int kt = 0; kt < 4; ++kt) {          // K = 128 = 4 tiles of 32
    v16bf a = aptr[(size_t)kt * 32];
#pragma unroll
    for (int nt = 0; nt < NT; ++nt) {
      v16bf b = bptr[(size_t)(kt * NT + nt) * 32];
      acc[nt] = __builtin_amdgcn_wmma_f32_16x16x32_bf16(
          /*neg_a=*/false, a, /*neg_b=*/false, b,
          /*c_mod=*/(short)0, acc[nt], /*reuse_a=*/false, /*reuse_b=*/false);
    }
  }

  // C/D layout: VGPR r -> row r (lanes 0-15) / row r+8 (lanes 16-31); col = lane&15.
  const int Dout = NT * 16;
  int half = lane >> 4;
  float* crow = C + (size_t)(mt * 16 + half * 8) * Dout + (lane & 15);
#pragma unroll
  for (int nt = 0; nt < NT; ++nt) {
#pragma unroll
    for (int r = 0; r < 8; ++r)
      crow[(size_t)r * Dout + nt * 16] = acc[nt][r];
  }
}

// ---------------------------------------------------------------- SpMM

// agg[dst[e]][:] += h[src[e]][:]   — one wave per edge, float4 per lane.
__global__ void spmm(const int* __restrict__ src, const int* __restrict__ dst,
                     const float* __restrict__ h, float* __restrict__ agg,
                     int nE, int D) {
  int wid  = (blockIdx.x * blockDim.x + threadIdx.x) >> 5;
  int lane = threadIdx.x & 31;
  if (wid >= nE) return;
  int s = src[wid];
  int d = dst[wid];
  int c = lane * 4;
  if (c < D) {
    const float4 v = *(const float4*)(h + (size_t)s * D + c);
    float* o = agg + (size_t)d * D + c;
    atomic_add_f32(o + 0, v.x);
    atomic_add_f32(o + 1, v.y);
    atomic_add_f32(o + 2, v.z);
    atomic_add_f32(o + 3, v.w);
  }
}

// ------------------------------------------- epilogues (fused, D = 128)

// y = relu(agg*in_norm + b); y = LN(y)*ln_s + ln_b; next A = bf16(y*out_norm)
// written directly in A-fragment order.  One wave per node, 4 dims/lane.
__global__ void finalize_ln(const float* __restrict__ agg, const float* __restrict__ in_norm,
                            const float* __restrict__ bias, const float* __restrict__ ln_s,
                            const float* __restrict__ ln_b, const float* __restrict__ out_norm,
                            bf16_t* __restrict__ apack_next, int nNodes) {
  int wid  = (blockIdx.x * blockDim.x + threadIdx.x) >> 5;
  int lane = threadIdx.x & 31;
  if (wid >= nNodes) return;

  const float4 v = ((const float4*)(agg + (size_t)wid * 128))[lane];
  float inn = in_norm[wid];
  int d0 = lane * 4;
  float y[4];
  y[0] = fmaxf(fmaf(v.x, inn, bias[d0 + 0]), 0.0f);
  y[1] = fmaxf(fmaf(v.y, inn, bias[d0 + 1]), 0.0f);
  y[2] = fmaxf(fmaf(v.z, inn, bias[d0 + 2]), 0.0f);
  y[3] = fmaxf(fmaf(v.w, inn, bias[d0 + 3]), 0.0f);

  float s  = y[0] + y[1] + y[2] + y[3];
  float ss = y[0]*y[0] + y[1]*y[1] + y[2]*y[2] + y[3]*y[3];
#pragma unroll
  for (int off = 16; off > 0; off >>= 1) {
    s  += __shfl_xor(s,  off, 32);
    ss += __shfl_xor(ss, off, 32);
  }
  float mean = s * (1.0f / 128.0f);
  float var  = ss * (1.0f / 128.0f) - mean * mean;
  float inv  = rsqrtf(var + 1e-5f);
  float onrm = out_norm[wid];

  int mt = wid >> 4;
  int ml = wid & 15;
#pragma unroll
  for (int j = 0; j < 4; ++j) {
    int d = d0 + j;
    float outv = fmaf((y[j] - mean) * inv, ln_s[d], ln_b[d]) * onrm;
    // invert A-fragment mapping: d -> (kt, half, i)
    int kt   = d >> 5;
    int k32  = d & 31;
    int half = (k32 >> 3) & 1;
    int kk   = k32 - half * 8;
    int i    = ((kk & 16) ? 8 : 0) + (kk & 7);
    apack_next[((size_t)((mt * 4 + kt) * 32 + half * 16 + ml)) * 16 + i] = (bf16_t)outv;
  }
}

// last layer: out = relu(agg*in_norm + b2), D = 64, fp32 output
__global__ void finalize_out(const float* __restrict__ agg, const float* __restrict__ in_norm,
                             const float* __restrict__ bias, float* __restrict__ out, int total) {
  int i = blockIdx.x * blockDim.x + threadIdx.x;
  if (i >= total) return;
  int n = i >> 6;
  int d = i & 63;
  out[i] = fmaxf(fmaf(agg[i], in_norm[n], bias[d]), 0.0f);
}

// ---------------------------------------------------------------- driver

extern "C" void kernel_launch(void* const* d_in, const int* in_sizes, int n_in,
                              void* d_out, int out_size, void* d_ws, size_t ws_size,
                              hipStream_t stream) {
  (void)in_sizes; (void)n_in; (void)out_size; (void)ws_size;

  const float* x    = (const float*)d_in[0];
  const int*   src  = (const int*)d_in[1];
  const int*   dst  = (const int*)d_in[2];
  const float* W0   = (const float*)d_in[3];
  const float* b0   = (const float*)d_in[4];
  const float* W1   = (const float*)d_in[5];
  const float* b1   = (const float*)d_in[6];
  const float* W2   = (const float*)d_in[7];
  const float* b2   = (const float*)d_in[8];
  const float* ln_s = (const float*)d_in[9];
  const float* ln_b = (const float*)d_in[10];
  float* out = (float*)d_out;

  char* w = (char*)d_ws;
  auto take = [&](size_t bytes) -> char* {
    char* p = w;
    w += (bytes + 255) & ~(size_t)255;
    return p;
  };
  float*  norm_out = (float*)take((size_t)GCN_N_NODES * 4);
  float*  norm_in  = (float*)take((size_t)GCN_N_NODES * 4);
  bf16_t* apack    = (bf16_t*)take((size_t)GCN_N_NODES * 128 * 2);
  bf16_t* wpack    = (bf16_t*)take((size_t)128 * 128 * 2);
  float*  h_mm     = (float*)take((size_t)GCN_N_NODES * 128 * 4);
  float*  agg      = (float*)take((size_t)GCN_N_NODES * 128 * 4);

  const int B = 256;
  const int Mtiles = GCN_N_NODES / 16;   // 3125
  auto cdiv = [](long a, long b) { return (int)((a + b - 1) / b); };

  // ---- degree norms
  zero_f<<<cdiv(GCN_N_NODES, B), B, 0, stream>>>(norm_out, GCN_N_NODES);
  zero_f<<<cdiv(GCN_N_NODES, B), B, 0, stream>>>(norm_in, GCN_N_NODES);
  degree_kernel<<<cdiv(GCN_N_EDGES, B), B, 0, stream>>>(src, dst, norm_out, norm_in, GCN_N_EDGES);
  make_norms<<<cdiv(GCN_N_NODES, B), B, 0, stream>>>(norm_out, GCN_N_NODES);
  make_norms<<<cdiv(GCN_N_NODES, B), B, 0, stream>>>(norm_in, GCN_N_NODES);

  // ---- layer 0 (128 -> 128, LN)
  pack_x<<<cdiv((long)GCN_N_NODES * 128, B), B, 0, stream>>>(x, norm_out, apack, GCN_N_NODES * 128);
  pack_w<<<cdiv(128 * 128, B), B, 0, stream>>>(W0, wpack, 128, 128 * 128);
  gemm_wmma_strip<8><<<cdiv(Mtiles, 8), 256, 0, stream>>>(apack, wpack, h_mm, Mtiles);
  zero_f<<<cdiv((long)GCN_N_NODES * 128, B), B, 0, stream>>>(agg, GCN_N_NODES * 128);
  spmm<<<cdiv((long)GCN_N_EDGES * 32, B), B, 0, stream>>>(src, dst, h_mm, agg, GCN_N_EDGES, 128);
  finalize_ln<<<cdiv((long)GCN_N_NODES * 32, B), B, 0, stream>>>(
      agg, norm_in, b0, ln_s, ln_b, norm_out, apack, GCN_N_NODES);

  // ---- layer 1 (128 -> 128, LN)
  pack_w<<<cdiv(128 * 128, B), B, 0, stream>>>(W1, wpack, 128, 128 * 128);
  gemm_wmma_strip<8><<<cdiv(Mtiles, 8), 256, 0, stream>>>(apack, wpack, h_mm, Mtiles);
  zero_f<<<cdiv((long)GCN_N_NODES * 128, B), B, 0, stream>>>(agg, GCN_N_NODES * 128);
  spmm<<<cdiv((long)GCN_N_EDGES * 32, B), B, 0, stream>>>(src, dst, h_mm, agg, GCN_N_EDGES, 128);
  finalize_ln<<<cdiv((long)GCN_N_NODES * 32, B), B, 0, stream>>>(
      agg, norm_in, b1, ln_s, ln_b, norm_out, apack, GCN_N_NODES);

  // ---- layer 2 (128 -> 64, relu only)
  pack_w<<<cdiv(128 * 64, B), B, 0, stream>>>(W2, wpack, 64, 128 * 64);
  gemm_wmma_strip<4><<<cdiv(Mtiles, 8), 256, 0, stream>>>(apack, wpack, h_mm, Mtiles);
  zero_f<<<cdiv((long)GCN_N_NODES * 64, B), B, 0, stream>>>(agg, GCN_N_NODES * 64);
  spmm<<<cdiv((long)GCN_N_EDGES * 32, B), B, 0, stream>>>(src, dst, h_mm, agg, GCN_N_EDGES, 64);
  finalize_out<<<cdiv((long)GCN_N_NODES * 64, B), B, 0, stream>>>(
      agg, norm_in, b2, out, GCN_N_NODES * 64);
}
